// MPNNBackbone_33131377721479
// MI455X (gfx1250) — matrix operand order repaired and checked by reading, not compiled, not run on hardware
//
#include <hip/hip_runtime.h>
#include <hip/hip_bf16.h>

// ---------------------------------------------------------------------------
// MPNN backbone on gfx1250 (wave32):
//   - x and edge_attr mirrored to bf16 once (x: per layer, in-place by upd),
//   - message GEMM [E,160]x[160,64] and update GEMM [N,128]x[128,64] via
//     v_wmma_f32_16x16x32_bf16 (fp32 accumulate),
//   - scatter-add aggregation via hardware global_atomic_add_f32.
// ---------------------------------------------------------------------------

typedef __bf16 bf16_t;
typedef __attribute__((ext_vector_type(16))) __bf16 v16bf;
typedef __attribute__((ext_vector_type(8)))  __bf16 v8bf;
typedef __attribute__((ext_vector_type(8)))  float  v8f;

#define WAVES     8          // 256 threads = 8 wave32 per block
#define DFEAT     64
#define KPAD_MSG  160        // 144 (2D + edge_dim) padded to 5 * 32
#define KPAD_UPD  128        // 2D, exact multiple of 32

__device__ __forceinline__ unsigned short f2bf_bits(float f) {
  union { float f; unsigned u; } c; c.f = f;
  return (unsigned short)((c.u + 0x7FFFu + ((c.u >> 16) & 1u)) >> 16); // RNE
}
__device__ __forceinline__ bf16_t f2bf(float f) {
  union { unsigned short s; bf16_t b; } o; o.s = f2bf_bits(f);
  return o.b;
}

// ---- one-time preprocessing -----------------------------------------------

// Weights -> bf16 column-major [l][n][kpad] so each lane's B fragment is one
// contiguous 32B load (lane = column, 16 consecutive K per lane half, same order as A).
__global__ void __launch_bounds__(256)
conv_w_kernel(const float* __restrict__ msg_w,   // [3][144][64]
              const float* __restrict__ upd_w,   // [3][128][64]
              bf16_t* __restrict__ Wm,           // [3][64][KPAD_MSG]
              bf16_t* __restrict__ Wu)           // [3][64][KPAD_UPD]
{
  const int totM = 3 * DFEAT * KPAD_MSG;
  const int totU = 3 * DFEAT * KPAD_UPD;
  int i = blockIdx.x * blockDim.x + threadIdx.x;
  if (i < totM) {
    int l = i / (DFEAT * KPAD_MSG);
    int r = i % (DFEAT * KPAD_MSG);
    int n = r / KPAD_MSG;
    int k = r % KPAD_MSG;
    float v = (k < 144) ? msg_w[((size_t)l * 144 + k) * DFEAT + n] : 0.0f;
    Wm[i] = f2bf(v);
  } else if (i < totM + totU) {
    int j = i - totM;
    int l = j / (DFEAT * KPAD_UPD);
    int r = j % (DFEAT * KPAD_UPD);
    int n = r / KPAD_UPD;
    int k = r % KPAD_UPD;
    Wu[j] = f2bf(upd_w[((size_t)l * 128 + k) * DFEAT + n]);
  }
}

// fp32 -> bf16 mirror (used for edge_attr once and for x at layer 0).
__global__ void __launch_bounds__(256)
conv_bf16_kernel(const float* __restrict__ src, bf16_t* __restrict__ dst,
                 int total /* multiple of 4 */)
{
  int i = (blockIdx.x * blockDim.x + threadIdx.x) * 4;
  if (i < total) {
    float4 v = *(const float4*)(src + i);
    unsigned lo = ((unsigned)f2bf_bits(v.y) << 16) | f2bf_bits(v.x);
    unsigned hi = ((unsigned)f2bf_bits(v.w) << 16) | f2bf_bits(v.z);
    *(uint2*)(dst + i) = make_uint2(lo, hi);
  }
}

// ---- message pass: m = relu([x16[dst] | x16[src] | ea16] @ Wm + b);
//      agg[dst] += m (hardware fp32 atomics) -------------------------------
__global__ void __launch_bounds__(WAVES * 32)
msg_kernel(const bf16_t* __restrict__ x16,        // [N][64] bf16
           const long long* __restrict__ edge_index, // [2][E]
           const bf16_t* __restrict__ ea16,       // [E][16] bf16
           const bf16_t* __restrict__ Wm,         // [64][KPAD_MSG] col-major
           const float* __restrict__ bias,        // [64]
           float* __restrict__ agg,               // [N][64]
           int E)
{
  __shared__ __align__(32) bf16_t As[WAVES][16][KPAD_MSG];

  const int wave = threadIdx.x >> 5;
  const int lane = threadIdx.x & 31;
  const int col  = lane & 15;
  const int half = lane >> 4;
  const int rsub  = lane >> 3;   // 0..3
  const int chunk = lane & 7;    // 0..7 (16B chunks of a 128B bf16 row)
  const int ntiles = (E + 15) >> 4;

  for (int tile = blockIdx.x * WAVES + wave; tile < ntiles;
       tile += gridDim.x * WAVES) {
    const int e0 = tile << 4;

    // ---- gather A = [x_i | x_j | edge_attr | 0] -> LDS (pure b128 copies)
    #pragma unroll
    for (int rb = 0; rb < 16; rb += 4) {
      const int r = rb + rsub;
      int e = e0 + r; if (e >= E) e = E - 1;
      const int s = (int)edge_index[e];                 // src = row 0
      const int d = (int)edge_index[(size_t)E + e];     // dst = row 1
      *(uint4*)&As[wave][r][chunk * 8] =
          *(const uint4*)(x16 + (size_t)d * DFEAT + chunk * 8);
      *(uint4*)&As[wave][r][DFEAT + chunk * 8] =
          *(const uint4*)(x16 + (size_t)s * DFEAT + chunk * 8);
    }
    {   // edge_attr (32B/row) + zero K-pad (32B/row): one iteration each
      const int r  = lane >> 1;
      const int c2 = (lane & 1) * 8;
      int e = e0 + r; if (e >= E) e = E - 1;
      *(uint4*)&As[wave][r][2 * DFEAT + c2] =
          *(const uint4*)(ea16 + (size_t)e * 16 + c2);
      *(uint4*)&As[wave][r][144 + c2] = make_uint4(0u, 0u, 0u, 0u);
    }
    // destination node ids for the 8 M-rows this lane owns in C/D layout
    int mydst[8];
    #pragma unroll
    for (int v = 0; v < 8; ++v) {
      int e = e0 + half * 8 + v; if (e >= E) e = E - 1;
      mydst[v] = (int)edge_index[(size_t)E + e];
    }
    __builtin_amdgcn_wave_barrier();

    // ---- [16x160] x [160x64]: 5 K-steps x 4 N-tiles of bf16 WMMA ----
    v8f acc0 = {}, acc1 = {}, acc2 = {}, acc3 = {};
    const bf16_t* arow = &As[wave][col][0];
    #pragma unroll
    for (int kt = 0; kt < KPAD_MSG / 32; ++kt) {
      const int kb = kt * 32 + half * 16;
      v8bf alo = *(const v8bf*)(arow + kb);
      v8bf ahi = *(const v8bf*)(arow + kb + 8);
      v16bf a = __builtin_shufflevector(alo, ahi,
                 0,1,2,3,4,5,6,7,8,9,10,11,12,13,14,15);
      const bf16_t* wb = Wm + (size_t)col * KPAD_MSG + kt * 32 + half * 16;
      v16bf b0 = *(const v16bf*)(wb);
      v16bf b1 = *(const v16bf*)(wb + 16 * KPAD_MSG);
      v16bf b2 = *(const v16bf*)(wb + 32 * KPAD_MSG);
      v16bf b3 = *(const v16bf*)(wb + 48 * KPAD_MSG);
      acc0 = __builtin_amdgcn_wmma_f32_16x16x32_bf16(false, a, false, b0, (short)0, acc0, false, false);
      acc1 = __builtin_amdgcn_wmma_f32_16x16x32_bf16(false, a, false, b1, (short)0, acc1, false, false);
      acc2 = __builtin_amdgcn_wmma_f32_16x16x32_bf16(false, a, false, b2, (short)0, acc2, false, false);
      acc3 = __builtin_amdgcn_wmma_f32_16x16x32_bf16(false, a, false, b3, (short)0, acc3, false, false);
    }

    // ---- bias + relu + atomic scatter-add into agg[dst] ----
    const float b0 = bias[col],      b1 = bias[16 + col];
    const float b2 = bias[32 + col], b3 = bias[48 + col];
    #pragma unroll
    for (int v = 0; v < 8; ++v) {
      const int e = e0 + half * 8 + v;
      if (e < E) {
        float* row = agg + (size_t)mydst[v] * DFEAT;
        float m0 = acc0[v] + b0; m0 = m0 > 0.0f ? m0 : 0.0f;
        float m1 = acc1[v] + b1; m1 = m1 > 0.0f ? m1 : 0.0f;
        float m2 = acc2[v] + b2; m2 = m2 > 0.0f ? m2 : 0.0f;
        float m3 = acc3[v] + b3; m3 = m3 > 0.0f ? m3 : 0.0f;
        unsafeAtomicAdd(row + col,      m0);
        unsafeAtomicAdd(row + 16 + col, m1);
        unsafeAtomicAdd(row + 32 + col, m2);
        unsafeAtomicAdd(row + 48 + col, m3);
      }
    }
    __builtin_amdgcn_wave_barrier();
  }
}

// ---- update: x' = relu([x | agg] @ Wu + b); writes x16 in place (bf16) and,
//      on the last layer, fp32 into d_out. ---------------------------------
__global__ void __launch_bounds__(WAVES * 32)
upd_kernel(bf16_t* __restrict__ x16,          // [N][64] bf16 (read + write)
           const float* __restrict__ agg,     // [N][64]
           const bf16_t* __restrict__ Wu,     // [64][KPAD_UPD] col-major
           const float* __restrict__ bias,    // [64]
           float* __restrict__ xout32,        // nullptr or [N][64] fp32
           int N)
{
  __shared__ __align__(32) bf16_t As[WAVES][16][KPAD_UPD];

  const int wave = threadIdx.x >> 5;
  const int lane = threadIdx.x & 31;
  const int col  = lane & 15;
  const int half = lane >> 4;
  const int rsub  = lane >> 3;
  const int chunk = lane & 7;
  const int ntiles = (N + 15) >> 4;

  for (int tile = blockIdx.x * WAVES + wave; tile < ntiles;
       tile += gridDim.x * WAVES) {
    const int n0 = tile << 4;

    // x half: pure b128 copies of the bf16 mirror
    #pragma unroll
    for (int rb = 0; rb < 16; rb += 4) {
      const int r = rb + rsub;
      int n = n0 + r; if (n >= N) n = N - 1;
      *(uint4*)&As[wave][r][chunk * 8] =
          *(const uint4*)(x16 + (size_t)n * DFEAT + chunk * 8);
    }
    // agg half: fp32 -> bf16 (2 values per lane per row)
    for (int r = 0; r < 16; ++r) {
      int n = n0 + r; if (n >= N) n = N - 1;
      const float2 av = *(const float2*)(agg + (size_t)n * DFEAT + lane * 2);
      As[wave][r][DFEAT + lane * 2]     = f2bf(av.x);
      As[wave][r][DFEAT + lane * 2 + 1] = f2bf(av.y);
    }
    __builtin_amdgcn_wave_barrier();

    v8f acc0 = {}, acc1 = {}, acc2 = {}, acc3 = {};
    const bf16_t* arow = &As[wave][col][0];
    #pragma unroll
    for (int kt = 0; kt < KPAD_UPD / 32; ++kt) {
      const int kb = kt * 32 + half * 16;
      v8bf alo = *(const v8bf*)(arow + kb);
      v8bf ahi = *(const v8bf*)(arow + kb + 8);
      v16bf a = __builtin_shufflevector(alo, ahi,
                 0,1,2,3,4,5,6,7,8,9,10,11,12,13,14,15);
      const bf16_t* wb = Wu + (size_t)col * KPAD_UPD + kt * 32 + half * 16;
      v16bf b0 = *(const v16bf*)(wb);
      v16bf b1 = *(const v16bf*)(wb + 16 * KPAD_UPD);
      v16bf b2 = *(const v16bf*)(wb + 32 * KPAD_UPD);
      v16bf b3 = *(const v16bf*)(wb + 48 * KPAD_UPD);
      acc0 = __builtin_amdgcn_wmma_f32_16x16x32_bf16(false, a, false, b0, (short)0, acc0, false, false);
      acc1 = __builtin_amdgcn_wmma_f32_16x16x32_bf16(false, a, false, b1, (short)0, acc1, false, false);
      acc2 = __builtin_amdgcn_wmma_f32_16x16x32_bf16(false, a, false, b2, (short)0, acc2, false, false);
      acc3 = __builtin_amdgcn_wmma_f32_16x16x32_bf16(false, a, false, b3, (short)0, acc3, false, false);
    }

    // bias + relu, transpose through LDS (reuse As) for coalesced stores
    const float b0 = bias[col],      b1 = bias[16 + col];
    const float b2 = bias[32 + col], b3 = bias[48 + col];
    float* Ts = (float*)&As[wave][0][0];      // 16x64 fp32 = 4 KB footprint
    #pragma unroll
    for (int v = 0; v < 8; ++v) {
      const int m = half * 8 + v;
      float m0 = acc0[v] + b0; m0 = m0 > 0.0f ? m0 : 0.0f;
      float m1 = acc1[v] + b1; m1 = m1 > 0.0f ? m1 : 0.0f;
      float m2 = acc2[v] + b2; m2 = m2 > 0.0f ? m2 : 0.0f;
      float m3 = acc3[v] + b3; m3 = m3 > 0.0f ? m3 : 0.0f;
      Ts[m * DFEAT + col]      = m0;
      Ts[m * DFEAT + 16 + col] = m1;
      Ts[m * DFEAT + 32 + col] = m2;
      Ts[m * DFEAT + 48 + col] = m3;
    }
    __builtin_amdgcn_wave_barrier();
    for (int r = 0; r < 16; ++r) {
      const int n = n0 + r;
      if (n < N) {
        const float2 o = *(const float2*)(Ts + r * DFEAT + lane * 2);
        // bf16 mirror for the next layer (in place; tile owns these rows)
        unsigned pk = ((unsigned)f2bf_bits(o.y) << 16) | f2bf_bits(o.x);
        *(unsigned*)(x16 + (size_t)n * DFEAT + lane * 2) = pk;
        if (xout32) *(float2*)(xout32 + (size_t)n * DFEAT + lane * 2) = o;
      }
    }
    __builtin_amdgcn_wave_barrier();
  }
}

extern "C" void kernel_launch(void* const* d_in, const int* in_sizes, int n_in,
                              void* d_out, int out_size, void* d_ws, size_t ws_size,
                              hipStream_t stream) {
  const float*     x          = (const float*)d_in[0];
  const long long* edge_index = (const long long*)d_in[1];
  const float*     edge_attr  = (const float*)d_in[2];
  const float*     msg_w      = (const float*)d_in[3];
  const float*     msg_b      = (const float*)d_in[4];
  const float*     upd_w      = (const float*)d_in[5];
  const float*     upd_b      = (const float*)d_in[6];

  const int N = in_sizes[0] / DFEAT;
  const int E = in_sizes[1] / 2;

  // workspace: agg(f32) | x16(bf16) | ea16(bf16) | Wm(bf16) | Wu(bf16)
  float*  agg  = (float*)d_ws;
  bf16_t* x16  = (bf16_t*)(agg + (size_t)N * DFEAT);
  bf16_t* ea16 = x16 + (size_t)N * DFEAT;
  bf16_t* Wm   = ea16 + (size_t)E * 16;
  bf16_t* Wu   = Wm + (size_t)3 * DFEAT * KPAD_MSG;

  {   // one-time conversions
    const int totW = 3 * DFEAT * KPAD_MSG + 3 * DFEAT * KPAD_UPD;
    conv_w_kernel<<<(totW + 255) / 256, 256, 0, stream>>>(msg_w, upd_w, Wm, Wu);
    const int totEA = E * 16;
    conv_bf16_kernel<<<(totEA / 4 + 255) / 256, 256, 0, stream>>>(edge_attr, ea16, totEA);
    const int totX = N * DFEAT;
    conv_bf16_kernel<<<(totX / 4 + 255) / 256, 256, 0, stream>>>(x, x16, totX);
  }

  const int etiles = (E + 15) / 16;
  const int vtiles = (N + 15) / 16;
  int mblocks = (etiles + WAVES - 1) / WAVES; if (mblocks > 4096) mblocks = 4096;
  int ublocks = (vtiles + WAVES - 1) / WAVES; if (ublocks > 4096) ublocks = 4096;

  for (int l = 0; l < 3; ++l) {
    hipMemsetAsync(agg, 0, (size_t)N * DFEAT * sizeof(float), stream);
    msg_kernel<<<mblocks, WAVES * 32, 0, stream>>>(
        x16, edge_index, ea16,
        Wm + (size_t)l * DFEAT * KPAD_MSG, msg_b + (size_t)l * DFEAT, agg, E);
    upd_kernel<<<ublocks, WAVES * 32, 0, stream>>>(
        x16, agg,
        Wu + (size_t)l * DFEAT * KPAD_UPD, upd_b + (size_t)l * DFEAT,
        (l == 2) ? (float*)d_out : nullptr, N);
  }
}
